// SpectralConv1d_71270687310621
// MI455X (gfx1250) — matrix operand order repaired and checked by reading, not compiled, not run on hardware
//
#include <hip/hip_runtime.h>
#include <hip/hip_bf16.h>

// SpectralConv1d (Hartley-variant) for MI455X / gfx1250, wave32 + bf16 WMMA.
// B=32, Cin=Cout=128, N=8192, MODES=64.

#define NSP   8192
#define MODES 64
#define ROWS  4096   // B*Cin == B*Cout
#define KSPLIT 8

typedef __attribute__((ext_vector_type(16))) __bf16       v16bf;
typedef __attribute__((ext_vector_type(2)))  __bf16       v2bf;
typedef __attribute__((ext_vector_type(8)))  float        v8f;
typedef __attribute__((ext_vector_type(8)))  unsigned int v8u;

// f32x2 -> packed bf16x2 (RNE). Prefer the native packed convert when the
// toolchain exposes it (v_cvt_pk_bf16_f32); otherwise plain __bf16 casts.
static __device__ __forceinline__ unsigned pack2(float a, float b) {
#if __has_builtin(__builtin_amdgcn_cvt_pk_bf16_f32)
  v2bf v = __builtin_amdgcn_cvt_pk_bf16_f32(a, b);
  return __builtin_bit_cast(unsigned, v);
#else
  v2bf v;
  v[0] = (__bf16)a;
  v[1] = (__bf16)b;
  return __builtin_bit_cast(unsigned, v);
#endif
}
static __device__ __forceinline__ unsigned short f2bf(float f) {
  unsigned u = __float_as_uint(f);
  unsigned r = u + 0x7FFFu + ((u >> 16) & 1u);   // round-to-nearest-even
  return (unsigned short)(r >> 16);
}
static __device__ __forceinline__ v8f vzero() {
  v8f z;
#pragma unroll
  for (int j = 0; j < 8; ++j) z[j] = 0.0f;
  return z;
}
static __device__ __forceinline__ v8f wmma_bf16(v16bf a, v16bf b, v8f c) {
  // D = A(16x32 bf16) x B(32x16 bf16) + C(16x16 f32)
  return __builtin_amdgcn_wmma_f32_16x16x32_bf16(false, a, false, b, (short)0, c,
                                                 false, false);
}
// B-fragment pregenerated in lane-layout: one contiguous 32B chunk per lane.
static __device__ __forceinline__ v16bf load_bfrag(const unsigned short* p) {
  return __builtin_bit_cast(v16bf, *(const v8u*)p);
}
// A-fragment from f32 row-major memory, contiguous-k layout (same k order as the B fragments):
// lanes 0-15: K = klo+{0..15} in h0..h15 (klo=0)
// lanes16-31: same rows, klo=16. Caller passes p already offset by klo.
static __device__ __forceinline__ v16bf make_afrag(const float* p) {
  const float4* q = (const float4*)p;          // p is 32B aligned by construction
  float4 c0 = q[0], c1 = q[1], c2 = q[2], c3 = q[3];  // p[0..15]
  v8u u;
  u[0] = pack2(c0.x, c0.y); u[1] = pack2(c0.z, c0.w);
  u[2] = pack2(c1.x, c1.y); u[3] = pack2(c1.z, c1.w);
  u[4] = pack2(c2.x, c2.y); u[5] = pack2(c2.z, c2.w);
  u[6] = pack2(c3.x, c3.y); u[7] = pack2(c3.z, c3.w);
  return __builtin_bit_cast(v16bf, u);
}
// C/D 16x16 f32 layout: VGPR j -> row j (lanes 0-15) / row j+8 (lanes 16-31),
// col = lane&15.
static __device__ __forceinline__ void store_tile(float* out, int ldc, v8f acc,
                                                  int lane, float scale) {
  int col  = lane & 15;
  int rofs = (lane >> 4) << 3;
#pragma unroll
  for (int j = 0; j < 8; ++j) out[(size_t)(rofs + j) * ldc + col] = acc[j] * scale;
}

// Transform matrix entry M[k][n] of the reference's radix-2 "cas" butterfly
// recursion (NOT the symmetric DHT): product of 13 stage factors.
static __device__ float casM(unsigned k, unsigned n) {
  float r = 1.0f;
  unsigned kc = k, Nc = NSP;
  while (Nc > 1u) {
    unsigned half = Nc >> 1;
    if (n & 1u) {
      unsigned km = kc & (half - 1u);
      float th = 6.283185307179586f * ((float)km / (float)Nc);
      float s, c;
      __sincosf(th, &s, &c);
      float f = c + s;                 // cas(theta)
      if (kc >= half) f = -f;
      r *= f;
    }
    kc &= (half - 1u);
    n >>= 1u;
    Nc = half;
  }
  return r;
}

// ---- Precompute: stage-1 B operand, fragment layout [kt(256)][mt(4)][lane(32)][h(16)]
// element (K=n_spatial, N=m_mode) = M[m][n];  B layout: lane<16 -> K h+0, lane>=16 -> K h+16.
__global__ void gen_acas(unsigned short* __restrict__ A) {
  unsigned t    = blockIdx.x * 256u + threadIdx.x;   // < 524288
  unsigned h    = t & 15u;
  unsigned lane = (t >> 4) & 31u;
  unsigned fm   = t >> 9;
  unsigned mt   = fm & 3u;
  unsigned kt   = fm >> 2;                            // < 256
  unsigned m    = mt * 16u + (lane & 15u);
  unsigned loc  = ((lane >> 4) << 4) + h;             // 0..31
  unsigned n    = kt * 32u + loc;
  A[t] = f2bf(casM(m, n));
}
// ---- Precompute: stage-3 B operand, layout [kt2(2)][ntile(512)][lane(32)][h(16)]
// element (K=m_mode, N=k_spatial) = M[k][m].
__global__ void gen_bcas(unsigned short* __restrict__ Bc) {
  unsigned t     = blockIdx.x * 256u + threadIdx.x;  // < 524288
  unsigned h     = t & 15u;
  unsigned lane  = (t >> 4) & 31u;
  unsigned fm    = t >> 9;
  unsigned ntile = fm & 511u;
  unsigned kt2   = fm >> 9;                           // < 2
  unsigned kcol  = ntile * 16u + (lane & 15u);        // spatial output index
  unsigned m     = kt2 * 32u + ((lane >> 4) << 4) + h;
  Bc[t] = f2bf(casM(kcol, m));
}

// ---- Stage 1: S_part[p][r][m] = sum over K-slice of x[r,n]*M[m,n]
// 1024 waves: 128 row-tiles (32 rows) x 8 K-slices. The 8 waves of a block
// share the same K-slice p, so their cas B-fragment loads hit WGP$.
__global__ void stage1(const float* __restrict__ X,
                       const unsigned short* __restrict__ Ac,
                       float* __restrict__ part) {
  int tid  = threadIdx.x;
  int lane = tid & 31;
  int wid  = blockIdx.x * 8 + (tid >> 5);
  int rt   = wid & 127;
  int p    = wid >> 7;
  int r0   = rt * 32;
  v8f acc[2][4];
#pragma unroll
  for (int a = 0; a < 2; ++a)
#pragma unroll
    for (int b = 0; b < 4; ++b) acc[a][b] = vzero();
  int m15 = lane & 15;
  int klo = (lane >> 4) * 16;
  const float* x0 = X + (size_t)(r0 + m15) * NSP + klo;
  const float* x1 = X + (size_t)(r0 + 16 + m15) * NSP + klo;
  for (int it = 0; it < 32; ++it) {
    int kt = p * 32 + it;
    v16bf a0 = make_afrag(x0 + kt * 32);
    v16bf a1 = make_afrag(x1 + kt * 32);
#pragma unroll
    for (int mt = 0; mt < 4; ++mt) {
      v16bf b = load_bfrag(Ac + (((size_t)kt * 4 + mt) * 32 + lane) * 16);
      acc[0][mt] = wmma_bf16(a0, b, acc[0][mt]);
      acc[1][mt] = wmma_bf16(a1, b, acc[1][mt]);
    }
  }
  float* pb = part + (size_t)p * ROWS * MODES;
#pragma unroll
  for (int rr = 0; rr < 2; ++rr)
#pragma unroll
    for (int mt = 0; mt < 4; ++mt)
      store_tile(pb + (size_t)(r0 + rr * 16) * MODES + mt * 16, MODES,
                 acc[rr][mt], lane, 1.0f);
}

// ---- Reduce K-slices: S[r][m] = sum_p part[p][r][m]
__global__ void reduce_s(const float* __restrict__ part, float* __restrict__ S) {
  int i = blockIdx.x * 256 + threadIdx.x;  // < 262144
  float s = 0.0f;
#pragma unroll
  for (int p = 0; p < KSPLIT; ++p) s += part[(size_t)p * ROWS * MODES + i];
  S[i] = s;
}

// ---- Stage 2 (f32, LDS-tiled): mixed[b,o,m] = sum_i S[b,i,m]*w[i,o,m]
// grid = 64 modes x 8 o-chunks of 16.
__global__ void stage2(const float* __restrict__ S, const float* __restrict__ W,
                       float* __restrict__ mixed) {
  __shared__ float sS[ROWS];        // 16 KB : S[:, m]
  __shared__ float sW[128 * 16];    //  8 KB : w[:, o0:o0+16, m]
  int tid = threadIdx.x;
  int m   = blockIdx.x & 63;
  int o0  = (blockIdx.x >> 6) * 16;
  for (int idx = tid; idx < ROWS; idx += 256) sS[idx] = S[(size_t)idx * MODES + m];
  for (int idx = tid; idx < 2048; idx += 256) {
    int i = idx >> 4, o = idx & 15;
    sW[idx] = W[((size_t)i * 128 + o0 + o) * MODES + m];
  }
  __syncthreads();
#pragma unroll
  for (int e = 0; e < 2; ++e) {
    int bo = tid * 2 + e;             // 512 outputs per block
    int b = bo >> 4, oo = bo & 15;
    const float* sp = &sS[b * 128];
    float sum = 0.0f;
#pragma unroll 8
    for (int i = 0; i < 128; ++i) sum += sp[i] * sW[i * 16 + oo];
    mixed[((size_t)b * 128 + o0 + oo) * MODES + m] = sum;
  }
}

// ---- Stage 3: out[r,k] = (1/N) * sum_m mixed[r,m]*M[k,m]
// 16384 waves: 256 row-tiles (16) x 64 col-chunks (128). K = 64 = 2 WMMA steps.
// The 8 waves of a block share the same col-chunk cc, so the 16 KB of cas
// B-fragments they read are identical -> served from WGP$ instead of 8x L2.
__global__ void stage3(const float* __restrict__ mixed,
                       const unsigned short* __restrict__ Bc,
                       float* __restrict__ out) {
  int tid  = threadIdx.x;
  int lane = tid & 31;
  int wid  = blockIdx.x * 8 + (tid >> 5);
  int rt   = wid & 255;          // consecutive within a block
  int cc   = wid >> 8;           // shared within a block
  int r0 = rt * 16, c0 = cc * 128;
  v8f acc[8];
#pragma unroll
  for (int i = 0; i < 8; ++i) acc[i] = vzero();
  int m15 = lane & 15, klo = (lane >> 4) * 16;
  const float* mr = mixed + (size_t)(r0 + m15) * MODES + klo;
#pragma unroll
  for (int kt = 0; kt < 2; ++kt) {
    v16bf a = make_afrag(mr + kt * 32);
#pragma unroll
    for (int nt = 0; nt < 8; ++nt) {
      int ntile = (c0 >> 4) + nt;
      v16bf b = load_bfrag(Bc + (((size_t)kt * 512 + ntile) * 32 + lane) * 16);
      acc[nt] = wmma_bf16(a, b, acc[nt]);
    }
  }
  const float sc = 1.0f / 8192.0f;
#pragma unroll
  for (int nt = 0; nt < 8; ++nt)
    store_tile(out + (size_t)r0 * NSP + c0 + nt * 16, NSP, acc[nt], lane, sc);
}

extern "C" void kernel_launch(void* const* d_in, const int* in_sizes, int n_in,
                              void* d_out, int out_size, void* d_ws, size_t ws_size,
                              hipStream_t stream) {
  const float* x = (const float*)d_in[0];   // [32,128,8192] f32
  const float* w = (const float*)d_in[1];   // [128,128,64]  f32
  float* out = (float*)d_out;               // [32,128,8192] f32
  char* ws = (char*)d_ws;
  // Workspace layout (12 MB total):
  unsigned short* Ac   = (unsigned short*)(ws);                       // 1 MB
  unsigned short* Bc   = (unsigned short*)(ws + ((size_t)1 << 20));   // 1 MB
  float*          part = (float*)(ws + ((size_t)2 << 20));            // 8 MB
  float*          S    = (float*)(ws + ((size_t)10 << 20));           // 1 MB
  float*          mix  = (float*)(ws + ((size_t)11 << 20));           // 1 MB

  gen_acas<<<2048, 256, 0, stream>>>(Ac);
  gen_bcas<<<2048, 256, 0, stream>>>(Bc);
  stage1  <<<128,  256, 0, stream>>>(x, Ac, part);
  reduce_s<<<1024, 256, 0, stream>>>(part, S);
  stage2  <<<512,  256, 0, stream>>>(S, w, mix);
  stage3  <<<2048, 256, 0, stream>>>(mix, Bc, out);
}
